// Attention_85581518340520
// MI455X (gfx1250) — matrix operand order repaired and checked by reading, not compiled, not run on hardware
//
#include <hip/hip_runtime.h>
#include <math.h>

typedef __attribute__((ext_vector_type(16))) _Float16 v16h;
typedef __attribute__((ext_vector_type(8)))  _Float16 v8h;
typedef __attribute__((ext_vector_type(4)))  _Float16 v4h;
typedef __attribute__((ext_vector_type(8)))  float    v8f;
typedef __attribute__((ext_vector_type(4)))  float    v4f;

#define BB 2
#define SS 2048
#define EE 2048
#define HH 16
#define DD 128

// ---------------- f32 -> f16 conversion (4-wide) ----------------
__global__ void f32_to_f16_kernel(const float* __restrict__ in,
                                  _Float16* __restrict__ out, int n) {
  int i4 = (blockIdx.x * blockDim.x + threadIdx.x) * 4;
  if (i4 + 3 < n) {
    v4f v = *(const v4f*)(in + i4);
    v4h h;
    h.x = (_Float16)v.x; h.y = (_Float16)v.y;
    h.z = (_Float16)v.z; h.w = (_Float16)v.w;
    *(v4h*)(out + i4) = h;
  } else {
    for (int i = i4; i < n; i++) out[i] = (_Float16)in[i];
  }
}

// ---------------- RoPE in-place on f16 Q and K ----------------
// one thread per rotation pair (i, i+64) of one (b,s,h)
__global__ void rope_kernel(_Float16* __restrict__ q, _Float16* __restrict__ k) {
  int t = blockIdx.x * blockDim.x + threadIdx.x;   // B*S*H*64 = 2^22 threads
  int i = t & 63;
  int h = (t >> 6) & (HH - 1);
  int s = (t >> 10) & (SS - 1);
  int b = t >> 21;
  if (b >= BB) return;
  // inv_freq = 10000^(-2i/128) = 2^(-(2i/128)*log2(10000))
  float invf = exp2f(-((float)(2 * i) / (float)DD) * 13.287712379549449f);
  float f = (float)s * invf;
  float c = cosf(f), sn = sinf(f);
  size_t base = ((size_t)b * SS + s) * EE + (size_t)h * DD;
  float x1 = (float)q[base + i], x2 = (float)q[base + i + 64];
  q[base + i]      = (_Float16)(x1 * c - x2 * sn);
  q[base + i + 64] = (_Float16)(x2 * c + x1 * sn);
  x1 = (float)k[base + i]; x2 = (float)k[base + i + 64];
  k[base + i]      = (_Float16)(x1 * c - x2 * sn);
  k[base + i + 64] = (_Float16)(x2 * c + x1 * sn);
}

// ---------------- GEMM: C[M,N] = A[M,K] * W[N,K]^T + bias ----------------
// wave tile 64x64 (4 A-frags x 4 B-frags = 16 wmma/kstep),
// block 8 waves (2m x 4n) -> 128x256 block tile
template<int OUTF16>
__global__ __launch_bounds__(256) void gemm_bias_kernel(
    const _Float16* __restrict__ A, const _Float16* __restrict__ W,
    const float* __restrict__ bias, void* __restrict__ Cout,
    int M, int N, int K) {
  const int lane = threadIdx.x & 31;
  const int wave = threadIdx.x >> 5;     // 0..7
  const int wm = wave >> 2;              // 0..1
  const int wn = wave & 3;               // 0..3
  const int lo = lane & 15, hi = lane >> 4;
  const int m0 = blockIdx.x * 128 + wm * 64;
  const int n0 = blockIdx.y * 256 + wn * 64;

  v8f c[4][4];
  for (int i = 0; i < 4; i++)
    for (int j = 0; j < 4; j++) c[i][j] = (v8f)0.f;

  const _Float16* arow[4];
  const _Float16* wrow[4];
  #pragma unroll
  for (int i = 0; i < 4; i++)
    arow[i] = A + (size_t)(m0 + i * 16 + lo) * K + hi * 8;
  #pragma unroll
  for (int j = 0; j < 4; j++)
    wrow[j] = W + (size_t)(n0 + j * 16 + lo) * K + hi * 8;

  for (int kk = 0; kk < K; kk += 32) {
    v16h a[4], bf[4];
    #pragma unroll
    for (int i = 0; i < 4; i++) {
      v8h p0 = *(const v8h*)(arow[i] + kk);
      v8h p1 = *(const v8h*)(arow[i] + kk + 16);
      a[i] = __builtin_shufflevector(p0, p1, 0,1,2,3,4,5,6,7,8,9,10,11,12,13,14,15);
    }
    #pragma unroll
    for (int j = 0; j < 4; j++) {
      v8h p0 = *(const v8h*)(wrow[j] + kk);
      v8h p1 = *(const v8h*)(wrow[j] + kk + 16);
      bf[j] = __builtin_shufflevector(p0, p1, 0,1,2,3,4,5,6,7,8,9,10,11,12,13,14,15);
    }
    if (kk + 32 < K) {  // hint next k-step into cache (global_prefetch_b8)
      __builtin_prefetch(arow[0] + kk + 32);
      __builtin_prefetch(wrow[0] + kk + 32);
    }
    #pragma unroll
    for (int i = 0; i < 4; i++)
      #pragma unroll
      for (int j = 0; j < 4; j++)
        c[i][j] = __builtin_amdgcn_wmma_f32_16x16x32_f16(
            false, a[i], false, bf[j], (short)0, c[i][j], false, false);
  }

  for (int i = 0; i < 4; i++)
    for (int j = 0; j < 4; j++) {
      int n = n0 + j * 16 + lo;
      float bv = bias[n];
      #pragma unroll
      for (int r = 0; r < 8; r++) {
        int m = m0 + i * 16 + r + 8 * hi;
        float val = c[i][j][r] + bv;
        if (OUTF16) ((_Float16*)Cout)[(size_t)m * N + n] = (_Float16)val;
        else        ((float*)Cout)[(size_t)m * N + n] = val;
      }
    }
}

// ---------------- Flash attention (causal, online softmax) ----------------
// grid (S/64, B*H), block 128 (4 waves x 16 query rows each)
__global__ __launch_bounds__(128) void flash_kernel(
    const _Float16* __restrict__ Q, const _Float16* __restrict__ Kt,
    const _Float16* __restrict__ V, _Float16* __restrict__ O) {
  __shared__ _Float16 Vt[128][40];      // transposed V chunk: [d][key], padded
  __shared__ _Float16 Pw[4][16][40];    // per-wave P tile: [m][key], padded

  const int lane = threadIdx.x & 31;
  const int wave = threadIdx.x >> 5;
  const int lo = lane & 15, hi = lane >> 4;
  const int bh = blockIdx.y;
  const int b = bh >> 4, h = bh & 15;
  const int mbase = blockIdx.x * 64 + wave * 16;
  const size_t headoff = (size_t)b * SS * EE + (size_t)h * DD;

  // Q fragments: 16 rows x 128 d as 4 (16x32) A-frags
  v16h aq[4];
  {
    const _Float16* qrow = Q + headoff + (size_t)(mbase + lo) * EE;
    #pragma unroll
    for (int kd = 0; kd < 4; kd++) {
      int c0 = kd * 32 + hi * 8;
      v8h p0 = *(const v8h*)(qrow + c0);
      v8h p1 = *(const v8h*)(qrow + c0 + 16);
      aq[kd] = __builtin_shufflevector(p0, p1, 0,1,2,3,4,5,6,7,8,9,10,11,12,13,14,15);
    }
  }

  v8f o[8];
  for (int d = 0; d < 8; d++) o[d] = (v8f)0.f;
  float mr[8], lr[8];
  for (int r = 0; r < 8; r++) { mr[r] = -3.0e38f; lr[r] = 0.f; }
  const float scale = 0.08838834764831845f;   // 1/sqrt(128)
  const float L2E = 1.4426950408889634f;

  // V staging split: this thread owns key (tid>>2), d-range [(tid&3)*32, +32)
  const int vkey = threadIdx.x >> 2;
  const int vdb  = (threadIdx.x & 3) * 32;

  const int nChunks = (blockIdx.x + 1) * 2;   // causal: keys up to block end
  for (int ch = 0; ch < nChunks; ch++) {
    const int n0 = ch * 32;
    __syncthreads();
    // stage V chunk transposed: Vt[d][n]; 4 x B128 loads + b16 scatter per thread
    {
      const _Float16* vrow = V + headoff + (size_t)(n0 + vkey) * EE + vdb;
      v8h t0 = *(const v8h*)(vrow);
      v8h t1 = *(const v8h*)(vrow + 8);
      v8h t2 = *(const v8h*)(vrow + 16);
      v8h t3 = *(const v8h*)(vrow + 24);
      #pragma unroll
      for (int j = 0; j < 8; j++) {
        Vt[vdb + j][vkey]      = t0[j];
        Vt[vdb + 8 + j][vkey]  = t1[j];
        Vt[vdb + 16 + j][vkey] = t2[j];
        Vt[vdb + 24 + j][vkey] = t3[j];
      }
    }
    __syncthreads();

    // scores for two 16-key tiles
    v8f sf[2];
    #pragma unroll
    for (int t = 0; t < 2; t++) {
      v8f acc = (v8f)0.f;
      const _Float16* krow = Kt + headoff + (size_t)(n0 + t * 16 + lo) * EE;
      #pragma unroll
      for (int kd = 0; kd < 4; kd++) {
        v8h p0 = *(const v8h*)(krow + kd * 32 + hi * 8);
        v8h p1 = *(const v8h*)(krow + kd * 32 + hi * 8 + 16);
        v16h bf = __builtin_shufflevector(p0, p1, 0,1,2,3,4,5,6,7,8,9,10,11,12,13,14,15);
        acc = __builtin_amdgcn_wmma_f32_16x16x32_f16(
            false, aq[kd], false, bf, (short)0, acc, false, false);
      }
      sf[t] = acc;
    }

    // scale + causal mask + row max (reduce across 16-lane halves)
    float newm[8];
    #pragma unroll
    for (int r = 0; r < 8; r++) {
      int row = mbase + r + 8 * hi;
      float s0 = sf[0][r] * scale;
      float s1 = sf[1][r] * scale;
      s0 = ((n0 + lo) > row) ? -3.0e38f : s0;
      s1 = ((n0 + 16 + lo) > row) ? -3.0e38f : s1;
      sf[0][r] = s0; sf[1][r] = s1;
      float v = fmaxf(s0, s1);
      for (int msk = 1; msk < 16; msk <<= 1)
        v = fmaxf(v, __shfl_xor(v, msk, 32));
      newm[r] = fmaxf(mr[r], v);
    }
    // exp + row sum
    float rs[8];
    #pragma unroll
    for (int r = 0; r < 8; r++) {
      float p0 = (sf[0][r] <= -1.0e38f) ? 0.f : exp2f((sf[0][r] - newm[r]) * L2E);
      float p1 = (sf[1][r] <= -1.0e38f) ? 0.f : exp2f((sf[1][r] - newm[r]) * L2E);
      sf[0][r] = p0; sf[1][r] = p1;
      float v = p0 + p1;
      for (int msk = 1; msk < 16; msk <<= 1)
        v += __shfl_xor(v, msk, 32);
      rs[r] = v;
    }
    // rescale accumulators, update stats
    #pragma unroll
    for (int r = 0; r < 8; r++) {
      float alpha = exp2f((mr[r] - newm[r]) * L2E);
      lr[r] = lr[r] * alpha + rs[r];
      mr[r] = newm[r];
      for (int d = 0; d < 8; d++) o[d][r] *= alpha;
    }
    // P (C-layout) -> LDS (row-major) as f16
    #pragma unroll
    for (int r = 0; r < 8; r++) {
      int m = r + 8 * hi;
      Pw[wave][m][lo]      = (_Float16)sf[0][r];
      Pw[wave][m][16 + lo] = (_Float16)sf[1][r];
    }
    __syncthreads();

    // PV: A = P (16x32), B = Vt column block (32 keys x 16 d)
    v16h pa;
    {
      const _Float16* pr = &Pw[wave][lo][0];
      int k0 = hi * 8;
      v8h p0 = *(const v8h*)(pr + k0);
      v8h p1 = *(const v8h*)(pr + k0 + 16);
      pa = __builtin_shufflevector(p0, p1, 0,1,2,3,4,5,6,7,8,9,10,11,12,13,14,15);
    }
    #pragma unroll
    for (int d = 0; d < 8; d++) {
      const _Float16* vr = &Vt[d * 16 + lo][hi * 8];
      v8h p0 = *(const v8h*)vr;
      v8h p1 = *(const v8h*)(vr + 16);
      v16h bf = __builtin_shufflevector(p0, p1, 0,1,2,3,4,5,6,7,8,9,10,11,12,13,14,15);
      o[d] = __builtin_amdgcn_wmma_f32_16x16x32_f16(
          false, pa, false, bf, (short)0, o[d], false, false);
    }
  }

  // normalize and store f16 attention output
  for (int r = 0; r < 8; r++) {
    int m = mbase + r + 8 * hi;
    float inv = 1.f / lr[r];
    #pragma unroll
    for (int d = 0; d < 8; d++)
      O[headoff + (size_t)m * EE + d * 16 + lo] = (_Float16)(o[d][r] * inv);
  }
}

// ---------------- host launch ----------------
extern "C" void kernel_launch(void* const* d_in, const int* in_sizes, int n_in,
                              void* d_out, int out_size, void* d_ws, size_t ws_size,
                              hipStream_t stream) {
  (void)in_sizes; (void)n_in; (void)out_size; (void)ws_size;
  const float* x  = (const float*)d_in[0];
  const float* Wq = (const float*)d_in[1];
  const float* bq = (const float*)d_in[2];
  const float* Wk = (const float*)d_in[3];
  const float* bk = (const float*)d_in[4];
  const float* Wv = (const float*)d_in[5];
  const float* bv = (const float*)d_in[6];
  const float* Wp = (const float*)d_in[7];
  const float* bp = (const float*)d_in[8];
  float* out = (float*)d_out;

  const size_t NX = (size_t)BB * SS * EE;   // 8,388,608
  const size_t NW = (size_t)EE * EE;        // 4,194,304
  _Float16* ws  = (_Float16*)d_ws;
  _Float16* xh  = ws;
  _Float16* wqh = xh + NX;
  _Float16* wkh = wqh + NW;
  _Float16* wvh = wkh + NW;
  _Float16* wph = wvh + NW;
  _Float16* qh  = wph + NW;
  _Float16* kh  = qh + NX;
  _Float16* vh  = kh + NX;
  _Float16* oh  = vh + NX;

  const int thr = 256;
  f32_to_f16_kernel<<<(int)((NX / 4 + thr - 1) / thr), thr, 0, stream>>>(x,  xh,  (int)NX);
  f32_to_f16_kernel<<<(int)((NW / 4 + thr - 1) / thr), thr, 0, stream>>>(Wq, wqh, (int)NW);
  f32_to_f16_kernel<<<(int)((NW / 4 + thr - 1) / thr), thr, 0, stream>>>(Wk, wkh, (int)NW);
  f32_to_f16_kernel<<<(int)((NW / 4 + thr - 1) / thr), thr, 0, stream>>>(Wv, wvh, (int)NW);
  f32_to_f16_kernel<<<(int)((NW / 4 + thr - 1) / thr), thr, 0, stream>>>(Wp, wph, (int)NW);

  const int M = BB * SS;                    // 4096
  dim3 ggrid(M / 128, EE / 256);            // (32, 8)
  gemm_bias_kernel<1><<<ggrid, 256, 0, stream>>>(xh, wqh, bq, qh, M, EE, EE);
  gemm_bias_kernel<1><<<ggrid, 256, 0, stream>>>(xh, wkh, bk, kh, M, EE, EE);
  gemm_bias_kernel<1><<<ggrid, 256, 0, stream>>>(xh, wvh, bv, vh, M, EE, EE);

  const int nrope = BB * SS * HH * 64;      // 4,194,304
  rope_kernel<<<nrope / thr, thr, 0, stream>>>(qh, kh);

  flash_kernel<<<dim3(SS / 64, BB * HH), 128, 0, stream>>>(qh, kh, vh, oh);

  gemm_bias_kernel<0><<<ggrid, 256, 0, stream>>>(oh, wph, bp, out, M, EE, EE);
}
